// TrianguLangSheafLoss_88158498717794
// MI455X (gfx1250) — compile-verified, matrix operand order checked by reading, not run on hardware
//
#include <hip/hip_runtime.h>
#include <hip/hip_bf16.h>
#include <stdint.h>

typedef __attribute__((ext_vector_type(16))) _Float16 v16h;
typedef __attribute__((ext_vector_type(8)))  _Float16 v8h;
typedef __attribute__((ext_vector_type(4)))  _Float16 v4h;
typedef __attribute__((ext_vector_type(8)))  float    v8f;

#define B_     4
#define N_     3
#define HW_    16384
#define D_     128
#define A_CON  256
#define A_CYC  128
#define THRESH2 0.01f          /* (0.1)^2 : mind<0.1  <=>  d2<0.01 */
#define INV_T  (1.0f/0.07f)

#define CHUNK_PTS   2048
#define CHUNK_BYTES (CHUNK_PTS*3*4)   /* 24576 B per LDS stage */

// ---------------------------------------------------------------------------
// Async global->LDS staging (CDNA5 GLOBAL_LOAD_ASYNC_TO_LDS_B128, ASYNCcnt)
// ---------------------------------------------------------------------------
template<int NT>
__device__ __forceinline__ void stage_chunk(float* sh, const float* gsrc, int tid) {
  constexpr int PER = CHUNK_BYTES / NT;       // bytes per thread (96 or 192)
  __syncthreads();                            // previous readers done
  uint32_t lbase = (uint32_t)(uintptr_t)sh + (uint32_t)(tid * PER);
  uint32_t goff  = (uint32_t)(tid * PER);
#pragma unroll
  for (int i = 0; i < PER / 16; ++i) {
    asm volatile("global_load_async_to_lds_b128 %0, %1, %2"
                 :: "v"(lbase + i * 16), "v"(goff + i * 16), "s"(gsrc)
                 : "memory");
  }
  asm volatile("s_wait_asynccnt 0" ::: "memory");
  __syncthreads();                            // all waves' DMA visible
}

// ---------------------------------------------------------------------------
// K1: normalize features f32 -> f16, one wave per point (128 dims / 32 lanes)
// ---------------------------------------------------------------------------
__global__ void tls_normalize_kernel(const float* __restrict__ feat,
                                     _Float16* __restrict__ nf) {
  int lane = threadIdx.x & 31;
  int wv   = threadIdx.x >> 5;
  long pt  = (long)blockIdx.x * 8 + wv;       // < B*N*HW
  const float4 x = ((const float4*)(feat + pt * D_))[lane];
  float ss = x.x*x.x + x.y*x.y + x.z*x.z + x.w*x.w;
#pragma unroll
  for (int m = 16; m >= 1; m >>= 1) ss += __shfl_xor(ss, m, 32);
  float sc = 1.0f / fmaxf(sqrtf(ss), 1e-12f);
  v4h o;
  o[0] = (_Float16)(x.x * sc); o[1] = (_Float16)(x.y * sc);
  o[2] = (_Float16)(x.z * sc); o[3] = (_Float16)(x.w * sc);
  *(v4h*)(nf + pt * D_ + lane * 4) = o;
}

// ---------------------------------------------------------------------------
// K2: contrast NN: 256 anchors vs 16384 points (LDS-staged chunks)
// ---------------------------------------------------------------------------
__global__ void tls_nn_contrast_kernel(const float* __restrict__ pm,
                                       float* __restrict__ mind,
                                       int* __restrict__ pos) {
  __shared__ float sh[CHUNK_PTS * 3];
  int pair = blockIdx.x;                      // 0..7 : b*2 + (pv-1)
  int b = pair >> 1, pv = 1 + (pair & 1);
  int a = threadIdx.x;                        // 0..255
  const float* p0 = pm + (((size_t)b * N_ + 0)  * HW_) * 3;
  const float* pj = pm + (((size_t)b * N_ + pv) * HW_) * 3;
  int hw = a * (HW_ / A_CON);
  float ax = p0[hw*3+0], ay = p0[hw*3+1], az = p0[hw*3+2];
  float bd = 3.4e38f; int bi = 0;
  for (int c = 0; c < HW_ / CHUNK_PTS; ++c) {
    stage_chunk<256>(sh, pj + (size_t)c * CHUNK_PTS * 3, threadIdx.x);
    for (int p = 0; p < CHUNK_PTS; ++p) {
      float dx = sh[p*3+0]-ax, dy = sh[p*3+1]-ay, dz = sh[p*3+2]-az;
      float d2 = dx*dx + dy*dy + dz*dz;
      if (d2 < bd) { bd = d2; bi = c*CHUNK_PTS + p; }
    }
  }
  mind[pair*A_CON + a] = sqrtf(fmaxf(bd, 1e-12f));
  pos [pair*A_CON + a] = bi;
}

// ---------------------------------------------------------------------------
// K3: sim = an @ alln^T / T, per-row logsumexp (no max needed: |sim|<=14.3)
//     one wave per 16-anchor tile; v_wmma_f32_16x16x32_f16, K=128 in 4 steps
// ---------------------------------------------------------------------------
__global__ void tls_sim_lse_kernel(const _Float16* __restrict__ nf,
                                   float* __restrict__ lse) {
  int pair = blockIdx.x >> 2;
  int b = pair >> 1, pv = 1 + (pair & 1);
  int wave = threadIdx.x >> 5, lane = threadIdx.x & 31;
  int tile  = (blockIdx.x & 3) * 4 + wave;    // 0..15
  int abase = tile * 16;
  const _Float16* nf0 = nf + ((size_t)b * N_ + 0)  * HW_ * D_;
  const _Float16* nfj = nf + ((size_t)b * N_ + pv) * HW_ * D_;
  int rl   = lane & 15;
  int koff = (lane < 16) ? 0 : 8;             // 16-bit A/B fragment K split
  int ahw  = (abase + rl) * (HW_ / A_CON);

  v16h afr[4];
#pragma unroll
  for (int kc = 0; kc < 4; ++kc) {            // A: rows=anchors, K=dims
    const _Float16* pa = nf0 + (size_t)ahw * D_ + kc * 32 + koff;
    v8h lo = *(const v8h*)pa;
    v8h hi = *(const v8h*)(pa + 16);
#pragma unroll
    for (int t = 0; t < 8; ++t) { afr[kc][t] = lo[t]; afr[kc][8+t] = hi[t]; }
  }
  float s[8];
#pragma unroll
  for (int r = 0; r < 8; ++r) s[r] = 0.0f;

  for (int jb = 0; jb < HW_; jb += 16) {
    const _Float16* bc = nfj + (size_t)(jb + rl) * D_ + koff;
    v8f c;
#pragma unroll
    for (int r = 0; r < 8; ++r) c[r] = 0.0f;
#pragma unroll
    for (int kc = 0; kc < 4; ++kc) {          // B: cols=points j, K=dims
      const _Float16* pb = bc + kc * 32;
      v8h lo = *(const v8h*)pb;
      v8h hi = *(const v8h*)(pb + 16);
      v16h bfr;
#pragma unroll
      for (int t = 0; t < 8; ++t) { bfr[t] = lo[t]; bfr[8+t] = hi[t]; }
      c = __builtin_amdgcn_wmma_f32_16x16x32_f16(false, afr[kc], false, bfr,
                                                 (short)0, c, false, false);
    }
#pragma unroll
    for (int r = 0; r < 8; ++r) s[r] += __expf(c[r] * INV_T);
  }
  // reduce each lane's column-subset sum across its 16-lane half
#pragma unroll
  for (int r = 0; r < 8; ++r) {
#pragma unroll
    for (int m = 1; m <= 8; m <<= 1) s[r] += __shfl_xor(s[r], m, 32);
  }
  if (rl == 0) {                              // lane0: rows r, lane16: rows r+8
    int rbase = abase + ((lane >> 4) ? 8 : 0);
#pragma unroll
    for (int r = 0; r < 8; ++r) lse[pair * A_CON + rbase + r] = __logf(s[r]);
  }
}

// ---------------------------------------------------------------------------
// K4: per-anchor loss = -pos_sim + lse, masked sum / count, ok = cnt>=5
// ---------------------------------------------------------------------------
__global__ void tls_contrast_reduce_kernel(const _Float16* __restrict__ nf,
                                           const float* __restrict__ mind,
                                           const int* __restrict__ pos,
                                           const float* __restrict__ lse,
                                           float* __restrict__ cnum,
                                           int* __restrict__ cok) {
  __shared__ float rs[256];
  __shared__ float rc[256];
  int pair = blockIdx.x;
  int b = pair >> 1, pv = 1 + (pair & 1);
  int a = threadIdx.x;
  const _Float16* nf0 = nf + ((size_t)b * N_ + 0)  * HW_ * D_;
  const _Float16* nfj = nf + ((size_t)b * N_ + pv) * HW_ * D_;
  int hw = a * (HW_ / A_CON);
  int pj = pos[pair*A_CON + a];
  const _Float16* pa = nf0 + (size_t)hw * D_;
  const _Float16* pb = nfj + (size_t)pj * D_;
  float dot = 0.f;
  for (int d = 0; d < D_; ++d) dot += (float)pa[d] * (float)pb[d];
  float per = -dot * INV_T + lse[pair*A_CON + a];
  bool va = mind[pair*A_CON + a] < 0.1f;
  rs[a] = va ? per : 0.f;
  rc[a] = va ? 1.f : 0.f;
  __syncthreads();
  for (int st = 128; st >= 1; st >>= 1) {
    if (a < st) { rs[a] += rs[a+st]; rc[a] += rc[a+st]; }
    __syncthreads();
  }
  if (a == 0) {
    int cnt = (int)rc[0];
    bool ok = cnt >= 5;                       // vi/vj sums == HW >= 10 always
    cnum[pair] = ok ? rs[0] / (float)max(cnt, 1) : 0.f;
    cok[pair]  = ok ? 1 : 0;
  }
}

// ---------------------------------------------------------------------------
// K5: cycle consistency for triplet (0,1,2): 3-hop NN chain + f16 MSE
// ---------------------------------------------------------------------------
__global__ void tls_cycle_kernel(const float* __restrict__ pm,
                                 const _Float16* __restrict__ nf,
                                 float* __restrict__ ynum,
                                 int* __restrict__ yok) {
  __shared__ float sh[CHUNK_PTS * 3];
  __shared__ float rs[A_CYC];
  __shared__ float rc[A_CYC];
  int b = blockIdx.x;
  int a = threadIdx.x;                        // 0..127
  const float* p0 = pm + (((size_t)b * N_) + 0) * HW_ * 3;
  const float* p1 = pm + (((size_t)b * N_) + 1) * HW_ * 3;
  const float* p2 = pm + (((size_t)b * N_) + 2) * HW_ * 3;
  int hw = a * (HW_ / A_CYC);
  float ax = p0[hw*3], ay = p0[hw*3+1], az = p0[hw*3+2];
  // hop 1: anchor -> p1
  float bd = 3.4e38f; int bi = 0;
  for (int c = 0; c < HW_ / CHUNK_PTS; ++c) {
    stage_chunk<A_CYC>(sh, p1 + (size_t)c * CHUNK_PTS * 3, a);
    for (int p = 0; p < CHUNK_PTS; ++p) {
      float dx = sh[p*3]-ax, dy = sh[p*3+1]-ay, dz = sh[p*3+2]-az;
      float d2 = dx*dx+dy*dy+dz*dz;
      if (d2 < bd) { bd = d2; bi = c*CHUNK_PTS + p; }
    }
  }
  float m1 = bd; int i1 = bi;
  float jx = p1[i1*3], jy = p1[i1*3+1], jz = p1[i1*3+2];
  // hop 2: pjc -> p2
  bd = 3.4e38f; bi = 0;
  for (int c = 0; c < HW_ / CHUNK_PTS; ++c) {
    stage_chunk<A_CYC>(sh, p2 + (size_t)c * CHUNK_PTS * 3, a);
    for (int p = 0; p < CHUNK_PTS; ++p) {
      float dx = sh[p*3]-jx, dy = sh[p*3+1]-jy, dz = sh[p*3+2]-jz;
      float d2 = dx*dx+dy*dy+dz*dz;
      if (d2 < bd) { bd = d2; bi = c*CHUNK_PTS + p; }
    }
  }
  float m2 = bd; int i2 = bi;
  float kx = p2[i2*3], ky = p2[i2*3+1], kz = p2[i2*3+2];
  // hop 3: pkc -> the 128 anchor positions (staged in LDS)
  __syncthreads();
  sh[a*3] = ax; sh[a*3+1] = ay; sh[a*3+2] = az;
  __syncthreads();
  bd = 3.4e38f; bi = 0;
  for (int p = 0; p < A_CYC; ++p) {
    float dx = sh[p*3]-kx, dy = sh[p*3+1]-ky, dz = sh[p*3+2]-kz;
    float d2 = dx*dx+dy*dy+dz*dz;
    if (d2 < bd) { bd = d2; bi = p; }
  }
  float m3 = bd; int iret = bi;               // raw row index, as in reference
  bool val = (m1 < THRESH2) && (m2 < THRESH2) && (m3 < THRESH2);
  const _Float16* nf0 = nf + ((size_t)b * N_) * HW_ * D_;
  const _Float16* fa = nf0 + (size_t)hw   * D_;
  const _Float16* fr = nf0 + (size_t)iret * D_;
  float mse = 0.f;
  for (int d = 0; d < D_; ++d) {
    float df = (float)fr[d] - (float)fa[d];
    mse += df * df;
  }
  mse *= (1.0f / D_);
  rs[a] = val ? mse : 0.f;
  rc[a] = val ? 1.f : 0.f;
  __syncthreads();
  for (int st = 64; st >= 1; st >>= 1) {
    if (a < st) { rs[a] += rs[a+st]; rc[a] += rc[a+st]; }
    __syncthreads();
  }
  if (a == 0) {
    int cnt = (int)rc[0];
    bool ok = cnt >= 5;
    ynum[b] = ok ? rs[0] / (float)max(cnt, 1) : 0.f;
    yok[b]  = ok ? 1 : 0;
  }
}

// ---------------------------------------------------------------------------
// K6: scalar combine
// ---------------------------------------------------------------------------
__global__ void tls_finalize_kernel(const float* __restrict__ cnum,
                                    const int* __restrict__ cok,
                                    const float* __restrict__ ynum,
                                    const int* __restrict__ yok,
                                    float* __restrict__ out) {
  if (threadIdx.x == 0 && blockIdx.x == 0) {
    float c_total = 0.f;
    for (int pv = 0; pv < N_ - 1; ++pv) {
      float s = 0.f; int okc = 0;
      for (int b = 0; b < B_; ++b) { s += cnum[b*(N_-1)+pv]; okc += cok[b*(N_-1)+pv]; }
      c_total += (okc > 0) ? s / (float)okc : 0.f;
    }
    float lc = c_total / (float)(N_ - 1);
    float s = 0.f; int okc = 0;
    for (int b = 0; b < B_; ++b) { s += ynum[b]; okc += yok[b]; }
    float lcy = (okc > 0) ? s / (float)okc : 0.f;   // nt == 1
    out[0] = 0.05f * lc + 0.1f * lcy;
  }
}

// ---------------------------------------------------------------------------
extern "C" void kernel_launch(void* const* d_in, const int* in_sizes, int n_in,
                              void* d_out, int out_size, void* d_ws, size_t ws_size,
                              hipStream_t stream) {
  (void)in_sizes; (void)n_in; (void)out_size;
  const float* pm   = (const float*)d_in[1];   // pointmaps [B,N,H,W,3] f32
  const float* feat = (const float*)d_in[2];   // features  [B,N,H,W,D] f32
  float* out = (float*)d_out;

  char* ws = (char*)d_ws;
  _Float16* nf = (_Float16*)ws;
  size_t off = (size_t)B_ * N_ * HW_ * D_ * sizeof(_Float16);   // 50.3 MB
  float* mind = (float*)(ws + off); off += (size_t)B_*(N_-1)*A_CON*4;
  int*   pos  = (int*)  (ws + off); off += (size_t)B_*(N_-1)*A_CON*4;
  float* lse  = (float*)(ws + off); off += (size_t)B_*(N_-1)*A_CON*4;
  float* cnum = (float*)(ws + off); off += (size_t)B_*(N_-1)*4;
  int*   cok  = (int*)  (ws + off); off += (size_t)B_*(N_-1)*4;
  float* ynum = (float*)(ws + off); off += (size_t)B_*4;
  int*   yok  = (int*)  (ws + off); off += (size_t)B_*4;
  if (ws_size < off) return;   // ws too small (constant per run -> deterministic)

  tls_normalize_kernel      <<<B_*N_*HW_/8,   256, 0, stream>>>(feat, nf);
  tls_nn_contrast_kernel    <<<B_*(N_-1),     256, 0, stream>>>(pm, mind, pos);
  tls_sim_lse_kernel        <<<B_*(N_-1)*4,   128, 0, stream>>>(nf, lse);
  tls_contrast_reduce_kernel<<<B_*(N_-1),     256, 0, stream>>>(nf, mind, pos, lse, cnum, cok);
  tls_cycle_kernel          <<<B_,          A_CYC, 0, stream>>>(pm, nf, ynum, yok);
  tls_finalize_kernel       <<<1,              32, 0, stream>>>(cnum, cok, ynum, yok, out);
}
